// TransformerLayer_74629351735818
// MI455X (gfx1250) — compile-verified, matrix operand order checked by reading, not run on hardware
//
#include <hip/hip_runtime.h>
#include <hip/hip_bf16.h>
#include <math.h>

typedef _Float16 f16;
typedef _Float16 v16h __attribute__((ext_vector_type(16)));
typedef _Float16 v8h  __attribute__((ext_vector_type(8)));
typedef float    v8f  __attribute__((ext_vector_type(8)));

#define NHEADS 16
#define SEQ    2048
#define ROWS   4096   // B*T

// ---------------------------------------------------------------- WMMA helpers
// Per-lane fragment load for 16-bit WMMA A/B operands (16x32 tile, row-major rows of K).
// lane half h2 selects K in {h2*8 .. h2*8+7} and {16+h2*8 .. 16+h2*8+7}.
__device__ __forceinline__ v16h frag_ld(const f16* rowp, int h2) {
  v16h f;
  *(v8h*)&f        = *(const v8h*)(rowp + h2 * 8);
  *((v8h*)&f + 1)  = *(const v8h*)(rowp + 16 + h2 * 8);
  return f;
}
__device__ __forceinline__ v8f wmma16(v16h a, v16h b, v8f c) {
  return __builtin_amdgcn_wmma_f32_16x16x32_f16(false, a, false, b, (short)0, c, false, false);
}

// ---------------------------------------------------------------- TDM (Tensor Data Mover)
#if defined(__HIP_DEVICE_COMPILE__) && __has_builtin(__builtin_amdgcn_tensor_load_to_lds)
#define USE_TDM 1
#else
#define USE_TDM 0
#endif

#if USE_TDM
typedef unsigned int v4u __attribute__((ext_vector_type(4)));
typedef int          v4i __attribute__((ext_vector_type(4)));
typedef int          v8i __attribute__((ext_vector_type(8)));

__device__ __forceinline__ unsigned lds_off(const void* p) {
  // generic pointers into LDS carry the LDS byte offset in the low 32 bits
  return (unsigned)(unsigned long long)p;
}

// 2-D tile load: 128 rows x 32 f16, row stride K elems, LDS rows padded to 80B
// (pad_enable, pad_interval = 16 DWORDs (64B), pad_amount = 4 DWORDs (16B)).
__device__ __forceinline__ void tdm_load_tile(const f16* gbase, unsigned lds_byte, int K) {
  unsigned long long ga = (unsigned long long)gbase;
  v4u g0;
  g0.x = 1u;                                              // count=1, no gather
  g0.y = lds_byte;                                        // lds_addr
  g0.z = (unsigned)(ga & 0xffffffffu);                    // global_addr[31:0]
  g0.w = (unsigned)((ga >> 32) & 0x01ffffffu) | (2u << 30); // global_addr[56:32] | type=2
  v8i g1;
  g1[0] = (1 << 16) | (1 << 20) | (3 << 22) | (3 << 25);  // data_size=2B, pad_en, 16DW interval, 4DW pad
  g1[1] = (32 & 0xffff) << 16;                            // tensor_dim0 = 32 (lo), abar=0
  g1[2] = (128 & 0xffff) << 16;                           // tensor_dim0 hi=0 | tensor_dim1 = 128 (lo)
  g1[3] = (32 & 0xffff) << 16;                            // tensor_dim1 hi=0 | tile_dim0 = 32
  g1[4] = (128 & 0xffff);                                 // tile_dim1 = 128 | tile_dim2 = 0
  g1[5] = K;                                              // tensor_dim0_stride[31:0]
  g1[6] = 0;                                              // stride hi | dim1_stride lo (unused for 2D)
  g1[7] = 0;
  v4i gz = {0, 0, 0, 0};
#if defined(__clang_major__) && (__clang_major__ >= 23)
  v8i gz8 = {0, 0, 0, 0, 0, 0, 0, 0};
  __builtin_amdgcn_tensor_load_to_lds(g0, g1, gz, gz, gz8, 0);
#else
  __builtin_amdgcn_tensor_load_to_lds(g0, g1, gz, gz, 0);
#endif
}
#endif

// ---------------------------------------------------------------- f32 -> f16
__global__ void k_cvt_f16(const float* __restrict__ in, f16* __restrict__ out, int n) {
  int i = blockIdx.x * 256 + threadIdx.x;
  if (i < n) out[i] = (f16)in[i];
}

// -------------------------------------------------- GEMM  C[M,N] = A[M,K] @ B[N,K]^T (+bias)
// block tile 128x128, BK=32, double-buffered LDS; 8 waves as 2x4 -> wave tile 64x32
// (8 WMMAs per 6 fragment loads per K-step)
__launch_bounds__(256)
__global__ void k_gemm(const f16* __restrict__ A, const f16* __restrict__ Bw,
                       const float* __restrict__ bias, float* __restrict__ C,
                       int M, int N, int K) {
  __shared__ __align__(16) f16 As[2][128 * 40];
  __shared__ __align__(16) f16 Bs[2][128 * 40];
  const int tid = threadIdx.x;
  const int lane = tid & 31, w = tid >> 5;
  const int wm = w >> 2, wn = w & 3;
  const int h2 = lane >> 4, ln = lane & 15;
  const int bM = blockIdx.y * 128, bN = blockIdx.x * 128;
  const int nk = K >> 5;

  v8f acc[4][2] = {};

#if USE_TDM
  if (w == 0) {
    tdm_load_tile(A + (size_t)bM * K, lds_off(&As[0][0]), K);
    tdm_load_tile(Bw + (size_t)bN * K, lds_off(&Bs[0][0]), K);
  }
  for (int i = 0; i < nk; ++i) {
    if (w == 0) {
      if (i + 1 < nk) {
        tdm_load_tile(A + (size_t)bM * K + (i + 1) * 32, lds_off(&As[(i + 1) & 1][0]), K);
        tdm_load_tile(Bw + (size_t)bN * K + (i + 1) * 32, lds_off(&Bs[(i + 1) & 1][0]), K);
        __builtin_amdgcn_s_wait_tensorcnt((short)2);   // tile i complete, tile i+1 in flight
      } else {
        __builtin_amdgcn_s_wait_tensorcnt((short)0);
      }
    }
    __syncthreads();
    const f16* as = As[i & 1];
    const f16* bs = Bs[i & 1];
    v16h af[4], bf[2];
    #pragma unroll
    for (int ti = 0; ti < 4; ++ti) af[ti] = frag_ld(&as[(wm * 64 + ti * 16 + ln) * 40], h2);
    #pragma unroll
    for (int tj = 0; tj < 2; ++tj) bf[tj] = frag_ld(&bs[(wn * 32 + tj * 16 + ln) * 40], h2);
    #pragma unroll
    for (int ti = 0; ti < 4; ++ti)
      #pragma unroll
      for (int tj = 0; tj < 2; ++tj)
        acc[ti][tj] = wmma16(af[ti], bf[tj], acc[ti][tj]);
    __syncthreads();   // all waves done with buf[i&1] before it is refilled
  }
#else
  // fallback: global -> regs -> LDS, double-buffered, one barrier per K-step
  const int ldr = tid & 127;                 // row within tile
  const bool isB = tid >= 128;
  const f16* gsrc = isB ? (Bw + (size_t)(bN + ldr) * K) : (A + (size_t)(bM + ldr) * K);
  {
    v8h r0 = *(const v8h*)(gsrc + 0),  r1 = *(const v8h*)(gsrc + 8);
    v8h r2 = *(const v8h*)(gsrc + 16), r3 = *(const v8h*)(gsrc + 24);
    f16* d = (isB ? Bs[0] : As[0]) + ldr * 40;
    *(v8h*)(d + 0) = r0; *(v8h*)(d + 8) = r1; *(v8h*)(d + 16) = r2; *(v8h*)(d + 24) = r3;
  }
  for (int i = 0; i < nk; ++i) {
    v8h r0, r1, r2, r3;
    if (i + 1 < nk) {
      const f16* g = gsrc + (i + 1) * 32;
      r0 = *(const v8h*)(g + 0);  r1 = *(const v8h*)(g + 8);
      r2 = *(const v8h*)(g + 16); r3 = *(const v8h*)(g + 24);
    }
    __syncthreads();                          // buf[i&1] ready for everyone
    const f16* as = As[i & 1];
    const f16* bs = Bs[i & 1];
    v16h af[4], bf[2];
    #pragma unroll
    for (int ti = 0; ti < 4; ++ti) af[ti] = frag_ld(&as[(wm * 64 + ti * 16 + ln) * 40], h2);
    #pragma unroll
    for (int tj = 0; tj < 2; ++tj) bf[tj] = frag_ld(&bs[(wn * 32 + tj * 16 + ln) * 40], h2);
    #pragma unroll
    for (int ti = 0; ti < 4; ++ti)
      #pragma unroll
      for (int tj = 0; tj < 2; ++tj)
        acc[ti][tj] = wmma16(af[ti], bf[tj], acc[ti][tj]);
    if (i + 1 < nk) {
      f16* d = (isB ? Bs[(i + 1) & 1] : As[(i + 1) & 1]) + ldr * 40;
      *(v8h*)(d + 0) = r0; *(v8h*)(d + 8) = r1; *(v8h*)(d + 16) = r2; *(v8h*)(d + 24) = r3;
    }
  }
#endif

  #pragma unroll
  for (int ti = 0; ti < 4; ++ti)
    #pragma unroll
    for (int tj = 0; tj < 2; ++tj) {
      int col = bN + wn * 32 + tj * 16 + ln;
      float bv = bias ? bias[col] : 0.0f;
      #pragma unroll
      for (int r = 0; r < 8; ++r) {
        int row = bM + wm * 64 + ti * 16 + r + 8 * h2;
        C[(size_t)row * N + col] = acc[ti][tj][r] + bv;
      }
    }
}

// ------------------------------------- RoPE + justnorm*sqk epilogue; V transposed to [bh][d][t]
__global__ void k_qkv_epi(const float* __restrict__ qkv, const float* __restrict__ sqk,
                          f16* __restrict__ Qh, f16* __restrict__ Kh, f16* __restrict__ Vt) {
  int lane = threadIdx.x & 31, w = threadIdx.x >> 5;
  int id = blockIdx.x * 8 + w;              // id = bh*SEQ + t
  int t  = id & (SEQ - 1);
  int bh = id >> 11;
  int h  = bh & (NHEADS - 1);
  int b  = bh >> 4;
  const float* base = qkv + ((size_t)b * SEQ + t) * 3072 + h * 192;

  // inv_freq[lane] = 10000^(-lane/32); ln(10000)/32 = 0.28782313662
  float fr = (float)t * __expf(-(float)lane * 0.28782313662425572f);
  float c = __cosf(fr), s = __sinf(fr);

  #pragma unroll
  for (int which = 0; which < 2; ++which) {           // 0 = Q, 1 = K
    const float* p = base + which * 64;
    float lo = p[lane], hi = p[lane + 32];
    float olo = lo * c - hi * s;                      // x*cos + rotate_half(x)*sin
    float ohi = hi * c + lo * s;
    float ss = olo * olo + ohi * ohi;
    #pragma unroll
    for (int o = 1; o < 32; o <<= 1) ss += __shfl_xor(ss, o);
    float inv = rsqrtf(ss);
    f16* q = (which == 0 ? Qh : Kh) + (size_t)id * 64;
    q[lane]      = (f16)(olo * inv * sqk[h * 64 + lane]);
    q[lane + 32] = (f16)(ohi * inv * sqk[h * 64 + lane + 32]);
  }
  Vt[((size_t)bh * 64 + lane) * SEQ + t]      = (f16)base[128 + lane];
  Vt[((size_t)bh * 64 + lane + 32) * SEQ + t] = (f16)base[128 + lane + 32];
}

// ---------------------------------------------------------- causal flash attention, WMMA
__launch_bounds__(128)
__global__ void k_attn(const f16* __restrict__ Qh, const f16* __restrict__ Kh,
                       const f16* __restrict__ Vt, float* __restrict__ Out) {
  __shared__ __align__(16) f16 PL[4][16 * 40];
  int lane = threadIdx.x & 31, w = threadIdx.x >> 5;
  int id = blockIdx.x * 4 + w;
  int qt = id & 127;                        // T/16 = 128 q-tiles
  int bh = id >> 7;
  int b = bh >> 4, h = bh & 15;
  int h2 = lane >> 4, ln = lane & 15;
  int qbase = qt * 16;

  const f16* Qrow = Qh + ((size_t)bh * SEQ + qbase + ln) * 64;
  v16h qf0 = frag_ld(Qrow, h2);             // d = 0..31
  v16h qf1 = frag_ld(Qrow + 32, h2);        // d = 32..63

  v8f o[4] = {};
  float m[8], l[8];
  #pragma unroll
  for (int r = 0; r < 8; ++r) { m[r] = -1e30f; l[r] = 0.0f; }

  f16* pl = PL[w];
  int nkt = (qt + 2) >> 1;                  // ceil((qbase+16)/32)
  for (int kb = 0; kb < nkt; ++kb) {
    int kbase = kb * 32;
    const f16* K0 = Kh + ((size_t)bh * SEQ + kbase + ln) * 64;
    const f16* K1 = K0 + 16 * 64;
    v16h k00 = frag_ld(K0, h2),      k01 = frag_ld(K0 + 32, h2);
    v16h k10 = frag_ld(K1, h2),      k11 = frag_ld(K1 + 32, h2);
    v8f s0 = {}, s1 = {};
    s0 = wmma16(qf0, k00, s0); s0 = wmma16(qf1, k01, s0);
    s1 = wmma16(qf0, k10, s1); s1 = wmma16(qf1, k11, s1);

    int col0 = kbase + ln, col1 = kbase + 16 + ln;
    #pragma unroll
    for (int r = 0; r < 8; ++r) {
      int row = qbase + r + 8 * h2;
      float v0 = (col0 <= row) ? s0[r] * 32.0f : -1e30f;   // scale = sqrt(1024)
      float v1 = (col1 <= row) ? s1[r] * 32.0f : -1e30f;
      float mx = fmaxf(v0, v1);
      mx = fmaxf(mx, __shfl_xor(mx, 1)); mx = fmaxf(mx, __shfl_xor(mx, 2));
      mx = fmaxf(mx, __shfl_xor(mx, 4)); mx = fmaxf(mx, __shfl_xor(mx, 8));
      float mn = fmaxf(m[r], mx);
      float al = __expf(m[r] - mn);
      m[r] = mn;
      float p0 = __expf(v0 - mn), p1 = __expf(v1 - mn);
      float rs = p0 + p1;
      rs += __shfl_xor(rs, 1); rs += __shfl_xor(rs, 2);
      rs += __shfl_xor(rs, 4); rs += __shfl_xor(rs, 8);
      l[r] = l[r] * al + rs;
      #pragma unroll
      for (int nt = 0; nt < 4; ++nt) o[nt][r] *= al;
      int rm = r + 8 * h2;
      pl[rm * 40 + ln]      = (f16)p0;
      pl[rm * 40 + 16 + ln] = (f16)p1;
    }
    asm volatile("s_wait_dscnt 0" ::: "memory");   // cross-lane LDS RAW inside the wave
    v16h ap = frag_ld(&pl[ln * 40], h2);           // P as 16x32 A fragment
    #pragma unroll
    for (int nt = 0; nt < 4; ++nt) {
      const f16* Vp = Vt + ((size_t)bh * 64 + nt * 16 + ln) * SEQ + kbase;
      v16h bv = frag_ld(Vp, h2);                   // V^T rows are contiguous in t
      o[nt] = wmma16(ap, bv, o[nt]);
    }
  }
  #pragma unroll
  for (int r = 0; r < 8; ++r) {
    float invl = 1.0f / l[r];
    int row = qbase + r + 8 * h2;
    size_t orow = ((size_t)b * SEQ + row) * 1024 + h * 64;
    #pragma unroll
    for (int nt = 0; nt < 4; ++nt)
      Out[orow + nt * 16 + ln] = o[nt][r] * invl;
  }
}

// ---------------------------------------------------------------- row justnorm -> f16
__global__ void k_rownorm_f16(const float* __restrict__ in, f16* __restrict__ out, int W) {
  __shared__ float red[256];
  int row = blockIdx.x, tid = threadIdx.x;
  const float* p = in + (size_t)row * W;
  float ss = 0.0f;
  for (int j = tid; j < W; j += 256) { float v = p[j]; ss += v * v; }
  red[tid] = ss; __syncthreads();
  for (int s = 128; s > 0; s >>= 1) { if (tid < s) red[tid] += red[tid + s]; __syncthreads(); }
  float inv = rsqrtf(red[0]);
  f16* q = out + (size_t)row * W;
  for (int j = tid; j < W; j += 256) q[j] = (f16)(p[j] * inv);
}

// --------------------------- out = justnorm(n(a) + |alpha*lrc| * (n(b) - n(a))), W = 1024
__global__ void k_residual(const float* __restrict__ a, const float* __restrict__ bb,
                           const float* __restrict__ alpha, float lrc,
                           float* __restrict__ outf, f16* __restrict__ outh) {
  __shared__ float red[256];
  const int W = 1024;
  int row = blockIdx.x, tid = threadIdx.x;
  const float* pa = a + (size_t)row * W;
  const float* pb = bb + (size_t)row * W;
  float sa = 0.0f, sb = 0.0f;
  #pragma unroll
  for (int u = 0; u < 4; ++u) {
    int j = tid + u * 256;
    float va = pa[j]; sa += va * va;
    float vb = pb[j]; sb += vb * vb;
  }
  red[tid] = sa; __syncthreads();
  for (int s = 128; s > 0; s >>= 1) { if (tid < s) red[tid] += red[tid + s]; __syncthreads(); }
  float na = rsqrtf(red[0]); __syncthreads();
  red[tid] = sb; __syncthreads();
  for (int s = 128; s > 0; s >>= 1) { if (tid < s) red[tid] += red[tid + s]; __syncthreads(); }
  float nb = rsqrtf(red[0]); __syncthreads();
  float y[4]; float sy = 0.0f;
  #pragma unroll
  for (int u = 0; u < 4; ++u) {
    int j = tid + u * 256;
    float An = pa[j] * na, Bn = pb[j] * nb;
    float lr = fabsf(alpha[j] * lrc);
    float v = An + lr * (Bn - An);
    y[u] = v; sy += v * v;
  }
  red[tid] = sy; __syncthreads();
  for (int s = 128; s > 0; s >>= 1) { if (tid < s) red[tid] += red[tid + s]; __syncthreads(); }
  float ny = rsqrtf(red[0]);
  #pragma unroll
  for (int u = 0; u < 4; ++u) {
    int j = tid + u * 256;
    float v = y[u] * ny;
    outf[(size_t)row * W + j] = v;
    if (outh) outh[(size_t)row * W + j] = (f16)v;
  }
}

// ------------------------- uv scale + justnorm(8192) + u*silu(v) -> f16 [4096]
__global__ void k_mlp_mid(const float* __restrict__ uv, const float* __restrict__ suv,
                          f16* __restrict__ out) {
  __shared__ float red[256];
  int row = blockIdx.x, tid = threadIdx.x;
  const float* p = uv + (size_t)row * 8192;
  float ss = 0.0f;
  for (int j = tid; j < 8192; j += 256) { float v = suv[j] * 32.0f * p[j]; ss += v * v; }
  red[tid] = ss; __syncthreads();
  for (int s = 128; s > 0; s >>= 1) { if (tid < s) red[tid] += red[tid + s]; __syncthreads(); }
  float inv = rsqrtf(red[0]);
  f16* q = out + (size_t)row * 4096;
  for (int j = tid; j < 4096; j += 256) {
    float u = suv[j] * 32.0f * p[j] * inv;
    float v = suv[j + 4096] * 32.0f * p[j + 4096] * inv;
    float sig = 1.0f / (1.0f + __expf(-v));
    q[j] = (f16)(u * v * sig);
  }
}

extern "C" void kernel_launch(void* const* d_in, const int* in_sizes, int n_in,
                              void* d_out, int out_size, void* d_ws, size_t ws_size,
                              hipStream_t stream) {
  (void)in_sizes; (void)n_in; (void)out_size; (void)ws_size;
  const float* x          = (const float*)d_in[0];
  const float* Wqkv       = (const float*)d_in[1];
  const float* sqk        = (const float*)d_in[2];
  const float* W_O        = (const float*)d_in[3];
  const float* attn_alpha = (const float*)d_in[4];
  const float* Wu_w       = (const float*)d_in[5];
  const float* Wu_b       = (const float*)d_in[6];
  const float* suv        = (const float*)d_in[7];
  const float* Wv_w       = (const float*)d_in[8];
  const float* Wv_b       = (const float*)d_in[9];
  const float* mlp_alpha  = (const float*)d_in[10];

  char* ws = (char*)d_ws;
  size_t off = 0;
  auto alloc = [&](size_t bytes) -> void* {
    void* p = ws + off; off += (bytes + 255) & ~(size_t)255; return p;
  };
  f16*   xh    = (f16*)  alloc((size_t)ROWS * 1024 * 2);
  f16*   Wqkvh = (f16*)  alloc((size_t)3072 * 1024 * 2);
  f16*   WOh   = (f16*)  alloc((size_t)1024 * 1024 * 2);
  f16*   Wuh   = (f16*)  alloc((size_t)8192 * 1024 * 2);
  f16*   Wvh   = (f16*)  alloc((size_t)1024 * 4096 * 2);
  float* qkv   = (float*)alloc((size_t)ROWS * 3072 * 4);
  f16*   Qh    = (f16*)  alloc((size_t)32 * SEQ * 64 * 2);
  f16*   Kh    = (f16*)  alloc((size_t)32 * SEQ * 64 * 2);
  f16*   Vt    = (f16*)  alloc((size_t)32 * 64 * SEQ * 2);
  float* t0    = (float*)alloc((size_t)ROWS * 1024 * 4);
  f16*   th    = (f16*)  alloc((size_t)ROWS * 1024 * 2);
  float* tproj = (float*)alloc((size_t)ROWS * 1024 * 4);
  float* x2    = (float*)alloc((size_t)ROWS * 1024 * 4);
  f16*   x2h   = (f16*)  alloc((size_t)ROWS * 1024 * 2);
  float* uv    = (float*)alloc((size_t)ROWS * 8192 * 4);
  f16*   resh  = (f16*)  alloc((size_t)ROWS * 4096 * 2);
  float* t2    = (float*)alloc((size_t)ROWS * 1024 * 4);

  auto cvt = [&](const float* in, f16* out, size_t n) {
    k_cvt_f16<<<dim3((unsigned)((n + 255) / 256)), 256, 0, stream>>>(in, out, (int)n);
  };
  cvt(x,    xh,    (size_t)ROWS * 1024);
  cvt(Wqkv, Wqkvh, (size_t)3072 * 1024);
  cvt(W_O,  WOh,   (size_t)1024 * 1024);
  cvt(Wu_w, Wuh,   (size_t)8192 * 1024);
  cvt(Wv_w, Wvh,   (size_t)1024 * 4096);

  // 1) qkv = x @ Wqkv^T   [4096 x 3072]
  k_gemm<<<dim3(3072 / 128, ROWS / 128), 256, 0, stream>>>(xh, Wqkvh, nullptr, qkv, ROWS, 3072, 1024);
  // 2) RoPE + norm + sqk, build Qh/Kh [bh][t][d] and Vt [bh][d][t]
  k_qkv_epi<<<dim3(32 * SEQ / 8), 256, 0, stream>>>(qkv, sqk, Qh, Kh, Vt);
  // 3) attention -> t0 [b][t][1024]
  k_attn<<<dim3(32 * 128 / 4), 128, 0, stream>>>(Qh, Kh, Vt, t0);
  // 4) th = justnorm(t0) as f16
  k_rownorm_f16<<<dim3(ROWS), 256, 0, stream>>>(t0, th, 1024);
  // 5) tproj = th @ W_O^T
  k_gemm<<<dim3(1024 / 128, ROWS / 128), 256, 0, stream>>>(th, WOh, nullptr, tproj, ROWS, 1024, 1024);
  // 6) x2 = justnorm(n(x) + |attn_alpha*1.6|*(n(tproj)-n(x)))   (0.05/C^-0.5 = 1.6)
  k_residual<<<dim3(ROWS), 256, 0, stream>>>(x, tproj, attn_alpha, 1.6f, x2, x2h);
  // 7) uv = x2 @ Wu^T + b  [4096 x 8192]
  k_gemm<<<dim3(8192 / 128, ROWS / 128), 256, 0, stream>>>(x2h, Wuh, Wu_b, uv, ROWS, 8192, 1024);
  // 8) resh = u * silu(v) of justnorm(suv*32*uv)
  k_mlp_mid<<<dim3(ROWS), 256, 0, stream>>>(uv, suv, resh);
  // 9) t2 = resh @ Wv^T + b  [4096 x 1024], K=4096
  k_gemm<<<dim3(1024 / 128, ROWS / 128), 256, 0, stream>>>(resh, Wvh, Wv_b, t2, ROWS, 1024, 4096);
  // 10) out = justnorm(x2 + |mlp_alpha*0.05|*(n(t2)-x2))
  k_residual<<<dim3(ROWS), 256, 0, stream>>>(x2, t2, mlp_alpha, 0.05f, (float*)d_out, nullptr);
}
